// TransformerBlock_3255585210707
// MI455X (gfx1250) — matrix operand order repaired and checked by reading, compiled, not run on hardware
//
#include <hip/hip_runtime.h>
#include <math.h>
#include <stdint.h>

typedef __attribute__((ext_vector_type(4)))  unsigned int u32x4;
typedef __attribute__((ext_vector_type(16))) __bf16       v16bf;
typedef __attribute__((ext_vector_type(8)))  float        v8f;

#define BB 2
#define SS 4096
#define DD 768
#define HH 12
#define DHH 64
#define MSROWS (BB * SS)   // 8192

static __device__ __forceinline__ v16bf frag_cat(u32x4 lo, u32x4 hi) {
  union { u32x4 p[2]; v16bf v; } u;
  u.p[0] = lo; u.p[1] = hi;
  return u.v;
}

static __device__ __forceinline__ v8f wmma_bf16(v16bf a, v16bf b, v8f c) {
  // 8 args: (neg_a, A, neg_b, B, c_mod, C, reuse_a, reuse_b)
  return __builtin_amdgcn_wmma_f32_16x16x32_bf16(false, a, false, b, (short)0, c,
                                                 false, false);
}

// CDNA5 async copy: global -> LDS, 16B per lane, tracked by ASYNCcnt.
static __device__ __forceinline__ void async_b128(__bf16* lds_dst,
                                                  const __bf16* gsrc) {
  unsigned l = (unsigned)(uintptr_t)lds_dst;  // flat->LDS: low 32 bits = LDS byte addr
  asm volatile("global_load_async_to_lds_b128 %0, %1, off"
               :: "v"(l), "v"(gsrc) : "memory");
}
static __device__ __forceinline__ void wait_async0() {
  asm volatile("s_wait_asynccnt 0x0" ::: "memory");
}

static __device__ __forceinline__ float gelu_f(float x) {
  float x3 = x * x * x;
  return 0.5f * x * (1.0f + tanhf(0.7978845608028654f * (x + 0.044715f * x3)));
}

// ---------------------------------------------------------------- conversions
__global__ __launch_bounds__(256) void k_f32_to_bf16(const float* __restrict__ in,
                                                     __bf16* __restrict__ out, int n) {
  int i = blockIdx.x * 256 + threadIdx.x;
  if (i < n) out[i] = (__bf16)in[i];
}

// in: f32 [K][N] row-major  ->  out: bf16 [N][K] row-major
__global__ __launch_bounds__(256) void k_transpose_bf16(const float* __restrict__ in,
                                                        __bf16* __restrict__ out,
                                                        int K, int N) {
  int i = blockIdx.x * 256 + threadIdx.x;
  if (i < K * N) {
    int k = i / N, n = i % N;
    out[(long)n * K + k] = (__bf16)in[i];
  }
}

// ---------------------------------------------------------------- WMMA GEMM
// C = act( A[MxK] @ W + bias ), W given transposed as BT[N][K] (bf16).
// Block: 128x128 tile, 8 waves, each wave 64x32 (4x2 WMMA tiles). K step 32.
// Double-buffered LDS filled with global_load_async_to_lds_b128.
__global__ __launch_bounds__(256) void k_gemm_bf16(
    const __bf16* __restrict__ A,    // M x K row-major
    const __bf16* __restrict__ BT,   // N x K row-major
    const float*  __restrict__ bias, // N
    float*  __restrict__ Cf,         // optional f32 output
    __bf16* __restrict__ Cb,         // optional bf16 output
    int M, int N, int K, int act_gelu) {
  __shared__ __attribute__((aligned(16))) __bf16 sA[2][128 * 32];
  __shared__ __attribute__((aligned(16))) __bf16 sB[2][128 * 32];
  (void)M;

  const int tid  = threadIdx.x;
  const int lane = tid & 31;
  const int wave = tid >> 5;
  const int wm = wave & 1;     // 64-row half
  const int wn = wave >> 1;    // 32-col slice
  const int ln = lane & 15;
  const int hi = lane >> 4;

  const long m0 = (long)blockIdx.y * 128;
  const long n0 = (long)blockIdx.x * 128;

  const int ldRow  = tid >> 1;         // 0..127
  const int ldHalf = (tid & 1) * 16;   // element offset in 32-wide row

  const __bf16* gA0 = A  + (m0 + ldRow) * (long)K + ldHalf;
  const __bf16* gB0 = BT + (n0 + ldRow) * (long)K + ldHalf;
  __bf16* dA[2] = { sA[0] + ldRow * 32 + ldHalf, sA[1] + ldRow * 32 + ldHalf };
  __bf16* dB[2] = { sB[0] + ldRow * 32 + ldHalf, sB[1] + ldRow * 32 + ldHalf };

  const v8f vzero = {0.f, 0.f, 0.f, 0.f, 0.f, 0.f, 0.f, 0.f};
  v8f acc[4][2];
#pragma unroll
  for (int i = 0; i < 4; ++i)
#pragma unroll
    for (int j = 0; j < 2; ++j) acc[i][j] = vzero;

  // prologue: stage first K tile into buffer 0
  async_b128(dA[0], gA0);
  async_b128(dA[0] + 8, gA0 + 8);
  async_b128(dB[0], gB0);
  async_b128(dB[0] + 8, gB0 + 8);

  int cur = 0;
  for (int k0 = 0; k0 < K; k0 += 32) {
    wait_async0();        // my share of the current tile has landed in LDS
    __syncthreads();      // everyone's share landed; prior reads of 'nxt' done

    if (k0 + 32 < K) {    // overlap next tile's async copy with this compute
      const int nxt = cur ^ 1;
      const __bf16* gA = gA0 + k0 + 32;
      const __bf16* gB = gB0 + k0 + 32;
      async_b128(dA[nxt], gA);
      async_b128(dA[nxt] + 8, gA + 8);
      async_b128(dB[nxt], gB);
      async_b128(dB[nxt] + 8, gB + 8);
    }

    v16bf af[4];
#pragma unroll
    for (int ms = 0; ms < 4; ++ms) {
      // A frag: lane L -> row L&15; 16 contiguous k at (L>>4)*16
      const __bf16* pa = sA[cur] + (wm * 64 + ms * 16 + ln) * 32 + hi * 16;
      af[ms] = frag_cat(*(const u32x4*)pa, *(const u32x4*)(pa + 8));
    }
#pragma unroll
    for (int ns = 0; ns < 2; ++ns) {
      // B frag: lane L -> column L&15; 16 contiguous k at (L>>4)*16
      const __bf16* pb = sB[cur] + (wn * 32 + ns * 16 + ln) * 32 + hi * 16;
      v16bf bf = frag_cat(((const u32x4*)pb)[0], ((const u32x4*)pb)[1]);
#pragma unroll
      for (int ms = 0; ms < 4; ++ms)
        acc[ms][ns] = wmma_bf16(af[ms], bf, acc[ms][ns]);
    }
    cur ^= 1;
  }

#pragma unroll
  for (int ms = 0; ms < 4; ++ms)
#pragma unroll
    for (int ns = 0; ns < 2; ++ns)
#pragma unroll
      for (int j = 0; j < 8; ++j) {
        long row = m0 + wm * 64 + ms * 16 + j + hi * 8;  // C layout: vgpr j -> M=j (+8 hi half)
        long col = n0 + wn * 32 + ns * 16 + ln;
        float v = acc[ms][ns][j] + bias[col];
        if (act_gelu) v = gelu_f(v);
        if (Cf) Cf[row * N + col] = v;
        if (Cb) Cb[row * N + col] = (__bf16)v;
      }
}

// ---------------------------------------------------------------- attention
// Flash attention: one wave per 16-query tile; 4 waves / block.
// qkv rows: [q(768) | k(768) | v(768)], per head 64 cols.
__global__ __launch_bounds__(128) void k_attn(
    const __bf16* __restrict__ qkv,        // MSROWS x 2304 bf16 (bias included)
    const unsigned char* __restrict__ pad, // B*S, 1 = padded key
    __bf16* __restrict__ obf) {            // MSROWS x 768 bf16
  __shared__ __attribute__((aligned(16))) __bf16 sP[4][16 * 32];
  __shared__ __attribute__((aligned(16))) __bf16 sVT[4][64 * 32];  // [dh][key]

  const int lane = threadIdx.x & 31;
  const int w    = threadIdx.x >> 5;
  const int ln = lane & 15;
  const int hi = lane >> 4;

  const int bh = blockIdx.x >> 6;   // 0..23  (SS/64 == 64 q-groups)
  const int qg = blockIdx.x & 63;
  const int b  = bh / HH;
  const int h  = bh % HH;
  const int q0 = (qg * 4 + w) * 16;
  const long rowbase = (long)b * SS;
  const int LDQ = 3 * DD;
  const float NEG = -1e30f;

  // Q fragments (16 x 64 bf16, two 32-wide k-chunks) straight from global
  v16bf qf[2];
  {
    const __bf16* qrow = qkv + (rowbase + q0 + ln) * LDQ + h * DHH + hi * 16;
#pragma unroll
    for (int c = 0; c < 2; ++c)
      qf[c] = frag_cat(*(const u32x4*)(qrow + c * 32),
                       *(const u32x4*)(qrow + c * 32 + 8));
  }

  const v8f vzero = {0.f, 0.f, 0.f, 0.f, 0.f, 0.f, 0.f, 0.f};
  v8f o[4];
#pragma unroll
  for (int os = 0; os < 4; ++os) o[os] = vzero;
  float m[8], l[8];
#pragma unroll
  for (int j = 0; j < 8; ++j) { m[j] = NEG; l[j] = 0.f; }

  const int ktiles = (q0 + 15) / 32 + 1;   // causal bound
  for (int kt = 0; kt < ktiles; ++kt) {
    const int kbase = kt * 32;

    // stage V tile (32 keys x 64 dh) transposed into sVT[dh][key]
    {
      const __bf16* vr = qkv + (rowbase + kbase + lane) * LDQ + 2 * DD + h * DHH;
      if (kt + 1 < ktiles)  // warm next key tile
        __builtin_prefetch((const void*)(vr + 32 * LDQ), 0, 3);
      union { u32x4 q[8]; __bf16 e[64]; } vv;
#pragma unroll
      for (int i = 0; i < 8; ++i) vv.q[i] = ((const u32x4*)vr)[i];
#pragma unroll
      for (int c = 0; c < 64; ++c) sVT[w][c * 32 + lane] = vv.e[c];
    }

    // scores S = Q @ K^T  (two 16-key column blocks)
    v8f s0 = vzero, s1 = vzero;
    {
      const __bf16* kr0 = qkv + (rowbase + kbase + ln) * LDQ      + DD + h * DHH + hi * 16;
      const __bf16* kr1 = qkv + (rowbase + kbase + 16 + ln) * LDQ + DD + h * DHH + hi * 16;
#pragma unroll
      for (int c = 0; c < 2; ++c) {
        v16bf k0f = frag_cat(((const u32x4*)(kr0 + c * 32))[0],
                             ((const u32x4*)(kr0 + c * 32))[1]);
        v16bf k1f = frag_cat(((const u32x4*)(kr1 + c * 32))[0],
                             ((const u32x4*)(kr1 + c * 32))[1]);
        s0 = wmma_bf16(qf[c], k0f, s0);
        s1 = wmma_bf16(qf[c], k1f, s1);
      }
    }

    const bool pad0 = pad[rowbase + kbase + ln]      != 0;
    const bool pad1 = pad[rowbase + kbase + 16 + ln] != 0;
    const int key0 = kbase + ln, key1 = kbase + 16 + ln;

    // online softmax; score row (j + 8*hi) lives across the 16-lane half group
#pragma unroll
    for (int j = 0; j < 8; ++j) {
      const int qr = q0 + j + hi * 8;
      float v0 = (!pad0 && key0 <= qr) ? s0[j] * 0.125f : NEG;
      float v1 = (!pad1 && key1 <= qr) ? s1[j] * 0.125f : NEG;
      float mt = fmaxf(v0, v1);
#pragma unroll
      for (int off = 1; off < 16; off <<= 1)
        mt = fmaxf(mt, __shfl_xor(mt, off, 32));
      const float nm = fmaxf(m[j], mt);
      const float sc = (m[j] > -1e29f) ? __expf(m[j] - nm) : 0.f;
      const float p0 = (v0 > -1e29f) ? __expf(v0 - nm) : 0.f;
      const float p1 = (v1 > -1e29f) ? __expf(v1 - nm) : 0.f;
      float ps = p0 + p1;
#pragma unroll
      for (int off = 1; off < 16; off <<= 1)
        ps += __shfl_xor(ps, off, 32);
      l[j] = l[j] * sc + ps;
      m[j] = nm;
#pragma unroll
      for (int os = 0; os < 4; ++os) o[os][j] *= sc;
      sP[w][(j + hi * 8) * 32 + ln]      = (__bf16)p0;
      sP[w][(j + hi * 8) * 32 + 16 + ln] = (__bf16)p1;
    }

    // O += P @ V
    {
      const __bf16* pr = sP[w] + ln * 32 + hi * 16;
      v16bf pf = frag_cat(*(const u32x4*)pr, *(const u32x4*)(pr + 8));
#pragma unroll
      for (int os = 0; os < 4; ++os) {
        const __bf16* vt = sVT[w] + (os * 16 + ln) * 32 + hi * 16;
        v16bf vf = frag_cat(((const u32x4*)vt)[0], ((const u32x4*)vt)[1]);
        o[os] = wmma_bf16(pf, vf, o[os]);
      }
    }
  }

  // normalize and write (all-masked rows -> 0, matching the reference)
#pragma unroll
  for (int os = 0; os < 4; ++os)
#pragma unroll
    for (int j = 0; j < 8; ++j) {
      const int qr = q0 + j + hi * 8;
      float val = (l[j] > 0.f) ? o[os][j] / l[j] : 0.f;
      obf[(rowbase + qr) * DD + h * DHH + os * 16 + ln] = (__bf16)val;
    }
}

// ---------------------------------------------------------------- layernorm
// Y = LN(X + R) * g + b ; one 256-thread block per 768-wide row.
__global__ __launch_bounds__(256) void k_ln(const float* __restrict__ X,
                                            const float* __restrict__ R,
                                            const float* __restrict__ g,
                                            const float* __restrict__ be,
                                            float* __restrict__ Yf,
                                            __bf16* __restrict__ Yb) {
  __shared__ float red[16];
  const long row = blockIdx.x;
  const int tid = threadIdx.x;
  const float* x = X + row * DD;
  const float* r = R + row * DD;
  float v[3];
  float s = 0.f;
#pragma unroll
  for (int i = 0; i < 3; ++i) {
    int c = tid + i * 256;
    v[i] = x[c] + r[c];
    s += v[i];
  }
#pragma unroll
  for (int off = 1; off < 32; off <<= 1) s += __shfl_xor(s, off, 32);
  if ((tid & 31) == 0) red[tid >> 5] = s;
  __syncthreads();
  float tot = 0.f;
#pragma unroll
  for (int i = 0; i < 8; ++i) tot += red[i];
  const float mean = tot * (1.0f / 768.0f);
  float vs = 0.f;
#pragma unroll
  for (int i = 0; i < 3; ++i) { float d = v[i] - mean; vs += d * d; }
#pragma unroll
  for (int off = 1; off < 32; off <<= 1) vs += __shfl_xor(vs, off, 32);
  if ((tid & 31) == 0) red[8 + (tid >> 5)] = vs;
  __syncthreads();
  float vtot = 0.f;
#pragma unroll
  for (int i = 0; i < 8; ++i) vtot += red[8 + i];
  const float rstd = rsqrtf(vtot * (1.0f / 768.0f) + 1e-5f);
#pragma unroll
  for (int i = 0; i < 3; ++i) {
    int c = tid + i * 256;
    float y = (v[i] - mean) * rstd * g[c] + be[c];
    Yf[row * DD + c] = y;
    if (Yb) Yb[row * DD + c] = (__bf16)y;
  }
}

// ---------------------------------------------------------------- launch
extern "C" void kernel_launch(void* const* d_in, const int* in_sizes, int n_in,
                              void* d_out, int out_size, void* d_ws, size_t ws_size,
                              hipStream_t stream) {
  (void)in_sizes; (void)n_in; (void)out_size; (void)ws_size;
  const float* x     = (const float*)d_in[0];
  const unsigned char* pad = (const unsigned char*)d_in[1];
  const float* qkv_w = (const float*)d_in[2];
  const float* qkv_b = (const float*)d_in[3];
  const float* out_w = (const float*)d_in[4];
  const float* out_b = (const float*)d_in[5];
  const float* ln1_g = (const float*)d_in[6];
  const float* ln1_b = (const float*)d_in[7];
  const float* w1    = (const float*)d_in[8];
  const float* b1    = (const float*)d_in[9];
  const float* w2    = (const float*)d_in[10];
  const float* b2    = (const float*)d_in[11];
  const float* ln2_g = (const float*)d_in[12];
  const float* ln2_b = (const float*)d_in[13];
  float* out = (float*)d_out;

  char* ws = (char*)d_ws;
  size_t off = 0;
  auto wsalloc = [&](size_t bytes) -> void* {
    void* p = ws + off;
    off = (off + bytes + 255) & ~(size_t)255;
    return p;
  };
  const long R = MSROWS;
  __bf16* xb   = (__bf16*)wsalloc((size_t)R * 768 * 2);
  __bf16* qkvb = (__bf16*)wsalloc((size_t)R * 2304 * 2);
  __bf16* obf  = (__bf16*)wsalloc((size_t)R * 768 * 2);
  float*  af   = (float*)wsalloc((size_t)R * 768 * 4);   // reused: attn-out, ffn-out
  float*  y1   = (float*)wsalloc((size_t)R * 768 * 4);
  __bf16* y1b  = (__bf16*)wsalloc((size_t)R * 768 * 2);
  __bf16* hb   = (__bf16*)wsalloc((size_t)R * 3072 * 2);
  __bf16* Wt1  = (__bf16*)wsalloc((size_t)768 * 2304 * 2);
  __bf16* Wt2  = (__bf16*)wsalloc((size_t)768 * 768 * 2);
  __bf16* Wt3  = (__bf16*)wsalloc((size_t)768 * 3072 * 2);
  __bf16* Wt4  = (__bf16*)wsalloc((size_t)3072 * 768 * 2);

  // prep: bf16 activations + transposed bf16 weights
  k_f32_to_bf16<<<dim3((unsigned)((R * 768 + 255) / 256)), dim3(256), 0, stream>>>(
      x, xb, (int)(R * 768));
  k_transpose_bf16<<<dim3((768 * 2304 + 255) / 256), dim3(256), 0, stream>>>(qkv_w, Wt1, 768, 2304);
  k_transpose_bf16<<<dim3((768 * 768 + 255) / 256),  dim3(256), 0, stream>>>(out_w, Wt2, 768, 768);
  k_transpose_bf16<<<dim3((768 * 3072 + 255) / 256), dim3(256), 0, stream>>>(w1, Wt3, 768, 3072);
  k_transpose_bf16<<<dim3((3072 * 768 + 255) / 256), dim3(256), 0, stream>>>(w2, Wt4, 3072, 768);

  // qkv = x @ qkv_w + qkv_b  -> bf16
  k_gemm_bf16<<<dim3(2304 / 128, (unsigned)(R / 128)), dim3(256), 0, stream>>>(
      xb, Wt1, qkv_b, nullptr, qkvb, (int)R, 2304, 768, 0);
  // flash attention -> bf16
  k_attn<<<dim3(BB * HH * (SS / 64)), dim3(128), 0, stream>>>(qkvb, pad, obf);
  // a = o @ out_w + out_b -> f32
  k_gemm_bf16<<<dim3(768 / 128, (unsigned)(R / 128)), dim3(256), 0, stream>>>(
      obf, Wt2, out_b, af, nullptr, (int)R, 768, 768, 0);
  // x1 = LN(x + a)
  k_ln<<<dim3((unsigned)R), dim3(256), 0, stream>>>(x, af, ln1_g, ln1_b, y1, y1b);
  // h = gelu(x1 @ w1 + b1) -> bf16
  k_gemm_bf16<<<dim3(3072 / 128, (unsigned)(R / 128)), dim3(256), 0, stream>>>(
      y1b, Wt3, b1, nullptr, hb, (int)R, 3072, 768, 1);
  // f = h @ w2 + b2 -> f32
  k_gemm_bf16<<<dim3(768 / 128, (unsigned)(R / 128)), dim3(256), 0, stream>>>(
      hb, Wt4, b2, af, nullptr, (int)R, 768, 3072, 0);
  // out = LN(x1 + f)
  k_ln<<<dim3((unsigned)R), dim3(256), 0, stream>>>(y1, af, ln2_g, ln2_b, out, nullptr);
}
